// CrossScaleAttention_4690104287732
// MI455X (gfx1250) — matrix operand order repaired and checked by reading, not compiled, not run on hardware
//
#include <hip/hip_runtime.h>

// ---------------- problem constants ----------------
#define NB   8
#define NC   128
#define NCM  64
#define NH   48
#define NW   48
#define NHW  2304      // 48*48
#define NHL  16
#define NL   256       // 16*16 reference patches
#define NK   576       // NCM*3*3
#define NOH  144       // 3*48
#define SMUL 10.0f

typedef __attribute__((ext_vector_type(16))) __bf16 v16bf;
typedef __attribute__((ext_vector_type(8)))  float  v8f;

// float -> bf16 round-to-nearest-even (prep/epilogue only)
__device__ __forceinline__ __bf16 f2bf(float f) {
    union { float f; unsigned u; } v; v.f = f;
    unsigned r = v.u + 0x7FFFu + ((v.u >> 16) & 1u);
    unsigned short h = (unsigned short)(r >> 16);
    return __builtin_bit_cast(__bf16, h);
}
__device__ __forceinline__ int b_kbase(int lane) { return (lane < 16) ? 0 : 16; }

// ================================================================
// prep kernels
// ================================================================
// x[b,c,p] f32 -> xbT[b,p,c] bf16  (channel-contiguous: feeds B fragments)
__global__ void k_cast_xT(const float* __restrict__ x, __bf16* __restrict__ xbT)
{
    size_t idx = (size_t)blockIdx.x * blockDim.x + threadIdx.x;
    if (idx >= (size_t)NB * NHW * NC) return;
    int c = (int)(idx % NC);
    size_t t = idx / NC;
    int p = (int)(t % NHW);
    int b = (int)(t / NHW);
    xbT[idx] = f2bf(x[((size_t)b * NC + c) * NHW + p]);
}

// weights [cout,128] f32 -> A-fragment-swizzled bf16:
// wz[((tm*4 + k/32)*32 + lane)*16 + j]
__global__ void k_wswz(const float* __restrict__ w, __bf16* __restrict__ wz, int cout)
{
    int idx = blockIdx.x * blockDim.x + threadIdx.x;
    if (idx >= cout * NC) return;
    int m = idx / NC, k = idx - m * NC;
    int lane = (m & 15) + 16 * ((k >> 4) & 1);
    int j    = k & 15;
    size_t dst = ((size_t)((m >> 4) * 4 + (k >> 5)) * 32 + lane) * 16 + j;
    wz[dst] = f2bf(w[idx]);
}

// ================================================================
// 1) 1x1 conv + bias + PReLU; one wave -> 16x16 tile; fragments are
//    single contiguous 32B vector loads (pre-swizzled W, transposed x)
// ================================================================
__global__ void k_conv1x1(const __bf16* __restrict__ xbT,  // [NB, NHW, NC]
                          const __bf16* __restrict__ wz,   // swizzled [cout/16][4][32][16]
                          const float* __restrict__ bias,
                          const float* __restrict__ alpha,
                          __bf16* __restrict__ y,          // [NB, cout, NHW]
                          int cout)
{
    int gw   = (blockIdx.x * blockDim.x + threadIdx.x) >> 5;
    int lane = threadIdx.x & 31;
    int tiles_n = NHW >> 4;
    int per_b = (cout >> 4) * tiles_n;
    if (gw >= NB * per_b) return;                 // wave-uniform exit
    int b  = gw / per_b;  int t  = gw - b * per_b;
    int tm = t / tiles_n; int tn = t - tm * tiles_n;

    int ncol = tn * 16 + (lane & 15);
    const __bf16* xrow = xbT + ((size_t)b * NHW + ncol) * NC + b_kbase(lane);
    const __bf16* wrow = wz + ((size_t)tm * 4) * 512 + lane * 16;

    v8f acc = {};
    #pragma unroll
    for (int ks = 0; ks < 4; ++ks) {
        v16bf af = *(const v16bf*)(wrow + ks * 512);
        v16bf bf = *(const v16bf*)(xrow + ks * 32);
        acc = __builtin_amdgcn_wmma_f32_16x16x32_bf16(false, af, false, bf,
                                                      (short)0, acc, false, false);
    }
    float al = alpha[0];
    int mbase = tm * 16 + ((lane < 16) ? 0 : 8);
    __bf16* yb = y + (size_t)b * cout * NHW;
    #pragma unroll
    for (int r = 0; r < 8; ++r) {
        int m = mbase + r;
        float v = acc[r] + bias[m];
        v = (v >= 0.f) ? v : al * v;
        yb[(size_t)m * NHW + ncol] = f2bf(v);
    }
}

// ================================================================
// 2) bilinear /3 (exact sample at 3i+1) + 1x1 conv -> ref f32 [NB,NCM,256]
// ================================================================
__global__ void k_ref_conv(const __bf16* __restrict__ xbT,
                           const __bf16* __restrict__ w2z, const float* __restrict__ b2,
                           const float* __restrict__ a2,
                           float* __restrict__ ref)
{
    int gw   = (blockIdx.x * blockDim.x + threadIdx.x) >> 5;
    int lane = threadIdx.x & 31;
    const int tiles_n = 256 >> 4;
    int per_b = (NCM >> 4) * tiles_n;
    if (gw >= NB * per_b) return;
    int b  = gw / per_b;  int t  = gw - b * per_b;
    int tm = t / tiles_n; int tn = t - tm * tiles_n;

    int ncol = tn * 16 + (lane & 15);
    int di = ncol >> 4, dj = ncol & 15;
    int sp = (3 * di + 1) * NW + (3 * dj + 1);
    const __bf16* xrow = xbT + ((size_t)b * NHW + sp) * NC + b_kbase(lane);
    const __bf16* wrow = w2z + ((size_t)tm * 4) * 512 + lane * 16;

    v8f acc = {};
    #pragma unroll
    for (int ks = 0; ks < 4; ++ks) {
        v16bf af = *(const v16bf*)(wrow + ks * 512);
        v16bf bf = *(const v16bf*)(xrow + ks * 32);
        acc = __builtin_amdgcn_wmma_f32_16x16x32_bf16(false, af, false, bf,
                                                      (short)0, acc, false, false);
    }
    float al = a2[0];
    int mbase = tm * 16 + ((lane < 16) ? 0 : 8);
    float* rb = ref + (size_t)b * NCM * 256;
    #pragma unroll
    for (int r = 0; r < 8; ++r) {
        int m = mbase + r;
        float v = acc[r] + b2[m];
        v = (v >= 0.f) ? v : al * v;
        rb[(size_t)m * 256 + ncol] = v;
    }
}

// ================================================================
// 3) wn: 3x3 patches of ref, per-patch L2 norm (clamp 1e-4) -> bf16 [b,l,576]
// ================================================================
__global__ void k_wn(const float* __restrict__ ref, __bf16* __restrict__ wn)
{
    int gw   = (blockIdx.x * blockDim.x + threadIdx.x) >> 5;
    int lane = threadIdx.x & 31;
    if (gw >= NB * NL) return;
    int b = gw / NL, l = gw - b * NL;
    int ph = l >> 4, pw = l & 15;
    const float* rb = ref + (size_t)b * NCM * 256;

    float vals[18];
    float sq = 0.f;
    #pragma unroll
    for (int t = 0; t < 18; ++t) {
        int k = lane + 32 * t;
        int c = k / 9, rem = k - c * 9;
        int ki = rem / 3, kj = rem - ki * 3;
        int ii = ph + ki - 1, jj = pw + kj - 1;
        float v = 0.f;
        if (ii >= 0 && ii < 16 && jj >= 0 && jj < 16)
            v = rb[(size_t)c * 256 + ii * 16 + jj];
        vals[t] = v;
        sq += v * v;
    }
    #pragma unroll
    for (int off = 16; off > 0; off >>= 1) sq += __shfl_xor(sq, off, 32);
    float inv = 1.f / fmaxf(sqrtf(sq), 1e-4f);
    __bf16* wb = wn + (size_t)gw * NK;
    #pragma unroll
    for (int t = 0; t < 18; ++t) wb[lane + 32 * t] = f2bf(vals[t] * inv);
}

// ================================================================
// 4) im2col of match, TRANSPOSED: xpT[b][p][k]  (k-contiguous rows)
// ================================================================
__global__ void k_im2colT(const __bf16* __restrict__ match, __bf16* __restrict__ xpT)
{
    size_t idx = (size_t)blockIdx.x * blockDim.x + threadIdx.x;
    if (idx >= (size_t)NB * NHW * NK) return;
    int k = (int)(idx % NK);
    size_t t = idx / NK;
    int p = (int)(t % NHW);
    int b = (int)(t / NHW);
    int c = k / 9, rem = k - c * 9;
    int ki = rem / 3, kj = rem - ki * 3;
    int h = p / NW, w = p - h * NW;
    int ii = h + ki - 1, jj = w + kj - 1;
    __bf16 v = f2bf(0.f);
    if (ii >= 0 && ii < NH && jj >= 0 && jj < NW)
        v = match[((size_t)b * NCM + c) * NHW + ii * NW + jj];
    xpT[idx] = v;
}

// ================================================================
// tiled GEMM machinery: block 256 thr, tile M64 x N128, Kstep 32.
// Both A (row-major) and B (n-major, K-contiguous) staged into LDS in
// WMMA *fragment layout* via async global->LDS b128 scatter copies, so every
// fragment read is one contiguous 32B LDS vector load.
// ================================================================
#define ALS_ELEMS (4 * 32 * 16)   // 4 A fragments (M subtiles)
#define BLS_ELEMS (8 * 32 * 16)   // 8 B fragments (N subtiles)

template<bool IS_A>
__device__ __forceinline__ void stage_chunk(const __bf16* base, long long ld,
                                            int row, int g, __bf16* lds)
{
    // row-chunk: 8 consecutive K elems (group g) of matrix row `row`
    const __bf16* src = base + (size_t)row * ld + 8 * g;
    int hs = g >> 1;     // lane-half selector
    int js = g & 1;     // j-half selector
    unsigned dst = (unsigned)(size_t)lds
                 + (unsigned)((row >> 4) * 1024 + ((row & 15) + 16 * hs) * 32 + js * 16);
    asm volatile("global_load_async_to_lds_b128 %0, %1, off"
                 :: "v"(dst), "v"(src) : "memory");
}

__device__ __forceinline__ void async_wait_barrier() {
    asm volatile("s_wait_asynccnt 0x0" ::: "memory");
    __syncthreads();
}

// ---- score GEMM: score[b] = wn[b](256x576) @ xpT[b]^T, f32 out ----
__global__ void __launch_bounds__(256)
k_score_t(const __bf16* __restrict__ wnp, const __bf16* __restrict__ xpT,
          float* __restrict__ score)
{
    __shared__ __bf16 Als[ALS_ELEMS];
    __shared__ __bf16 Bls[BLS_ELEMS];
    int tid = threadIdx.x, lane = tid & 31, w = tid >> 5;
    int wm = w >> 1, wnc = w & 1;

    int blk  = blockIdx.x;
    int nblk = blk % 18; blk /= 18;         // N: 2304/128
    int mblk = blk % 4;  int b = blk / 4;   // M: 256/64

    const __bf16* A  = wnp + ((size_t)b * NL + mblk * 64) * NK;
    const __bf16* BT = xpT + ((size_t)b * NHW + nblk * 128) * NK;

    v8f acc[4] = {};
    for (int k0 = 0; k0 < NK; k0 += 32) {
        __syncthreads();
        stage_chunk<true >(A  + k0, NK, tid >> 2,        tid & 3, Als);
        stage_chunk<false>(BT + k0, NK, tid >> 2,        tid & 3, Bls);
        stage_chunk<false>(BT + k0, NK, (tid >> 2) + 64, tid & 3, Bls);
        async_wait_barrier();
        v16bf a = *(const v16bf*)(Als + wm * 512 + lane * 16);
        #pragma unroll
        for (int nt = 0; nt < 4; ++nt) {
            v16bf bf = *(const v16bf*)(Bls + (wnc * 4 + nt) * 512 + lane * 16);
            acc[nt] = __builtin_amdgcn_wmma_f32_16x16x32_bf16(false, a, false, bf,
                                                              (short)0, acc[nt], false, false);
        }
    }
    float* sb = score + (size_t)b * NL * NHW;
    int mbase = mblk * 64 + wm * 16 + ((lane < 16) ? 0 : 8);
    #pragma unroll
    for (int nt = 0; nt < 4; ++nt) {
        int n = nblk * 128 + wnc * 64 + nt * 16 + (lane & 15);
        #pragma unroll
        for (int r = 0; r < 8; ++r)
            sb[(size_t)(mbase + r) * NHW + n] = acc[nt][r];
    }
}

// ================================================================
// 6) softmax over L=256 (scaled by 10), in place
// ================================================================
__global__ void k_softmax(float* __restrict__ score)
{
    int gw   = (blockIdx.x * blockDim.x + threadIdx.x) >> 5;
    int lane = threadIdx.x & 31;
    if (gw >= NB * NHW) return;
    int b = gw / NHW, p = gw - b * NHW;
    float* sc = score + (size_t)b * NL * NHW + p;

    float v[8];
    #pragma unroll
    for (int i = 0; i < 8; ++i) v[i] = sc[(size_t)(lane + 32 * i) * NHW] * SMUL;
    float mx = v[0];
    #pragma unroll
    for (int i = 1; i < 8; ++i) mx = fmaxf(mx, v[i]);
    #pragma unroll
    for (int off = 16; off > 0; off >>= 1) mx = fmaxf(mx, __shfl_xor(mx, off, 32));
    float sum = 0.f;
    #pragma unroll
    for (int i = 0; i < 8; ++i) { v[i] = __expf(v[i] - mx); sum += v[i]; }
    #pragma unroll
    for (int off = 16; off > 0; off >>= 1) sum += __shfl_xor(sum, off, 32);
    float inv = 1.f / sum;
    #pragma unroll
    for (int i = 0; i < 8; ++i) sc[(size_t)(lane + 32 * i) * NHW] = v[i] * inv;
}

// ================================================================
// 7) collapse attention, TRANSPOSED out: ScT[b][p][l']
// ================================================================
__global__ void k_collapseT(const float* __restrict__ attn, __bf16* __restrict__ ScT)
{
    size_t idx = (size_t)blockIdx.x * blockDim.x + threadIdx.x;
    if (idx >= (size_t)NB * NHW * NL) return;
    int l = (int)(idx % NL);
    size_t t = idx / NL;
    int p = (int)(t % NHW);
    int b = (int)(t / NHW);
    int php = l >> 4, pwp = l & 15;
    int q = p / NW, s = p - q * NW;
    const float* ab = attn + (size_t)b * NL * NHW;
    float sum = 0.f;
    #pragma unroll
    for (int u = 0; u < 3; ++u) {
        int ph = php + 1 - u, h = q + 1 - u;
        if (ph < 0 || ph >= NHL || h < 0 || h >= NH) continue;
        #pragma unroll
        for (int v = 0; v < 3; ++v) {
            int pw = pwp + 1 - v, ww = s + 1 - v;
            if (pw < 0 || pw >= NHL || ww < 0 || ww >= NW) continue;
            sum += ab[(size_t)(ph * NHL + pw) * NHW + h * NW + ww];
        }
    }
    ScT[idx] = f2bf(sum);
}

// ================================================================
// 8a) pixel-unshuffle repack: Ep[b][rt][c][l'] = embed[b,c,3ph+r,3pw+t]
// ================================================================
__global__ void k_epack(const __bf16* __restrict__ embed, __bf16* __restrict__ Ep)
{
    size_t idx = (size_t)blockIdx.x * blockDim.x + threadIdx.x;
    if (idx >= (size_t)NB * 9 * NC * NL) return;
    int l  = (int)(idx % NL);
    size_t t = idx / NL;
    int c  = (int)(t % NC); t /= NC;
    int rt = (int)(t % 9);
    int b  = (int)(t / 9);
    int r = rt / 3, tt = rt - 3 * r;
    int ph = l >> 4, pw = l & 15;
    Ep[idx] = embed[((size_t)b * NC + c) * NHW + (3 * ph + r) * NW + (3 * pw + tt)];
}

// ---- 8b) tconv GEMM: per (b,rt): out_rt = Ep(128x256) @ ScT^T / 6 ----
__global__ void __launch_bounds__(256)
k_tconv_t(const __bf16* __restrict__ Ep, const __bf16* __restrict__ ScT,
          float* __restrict__ out)
{
    __shared__ __bf16 Als[ALS_ELEMS];
    __shared__ __bf16 Bls[BLS_ELEMS];
    int tid = threadIdx.x, lane = tid & 31, w = tid >> 5;
    int wm = w >> 1, wnc = w & 1;

    int blk  = blockIdx.x;
    int nblk = blk % 18; blk /= 18;          // N: 2304/128
    int mblk = blk % 2;  blk /= 2;           // M: 128/64
    int rt   = blk % 9;  int b = blk / 9;
    int r = rt / 3, tt = rt - 3 * r;

    const __bf16* A  = Ep + (((size_t)b * 9 + rt) * NC + mblk * 64) * NL;
    const __bf16* BT = ScT + ((size_t)b * NHW + nblk * 128) * NL;

    v8f acc[4] = {};
    #pragma unroll
    for (int k0 = 0; k0 < NL; k0 += 32) {
        __syncthreads();
        stage_chunk<true >(A  + k0, NL, tid >> 2,        tid & 3, Als);
        stage_chunk<false>(BT + k0, NL, tid >> 2,        tid & 3, Bls);
        stage_chunk<false>(BT + k0, NL, (tid >> 2) + 64, tid & 3, Bls);
        async_wait_barrier();
        v16bf a = *(const v16bf*)(Als + wm * 512 + lane * 16);
        #pragma unroll
        for (int nt = 0; nt < 4; ++nt) {
            v16bf bf = *(const v16bf*)(Bls + (wnc * 4 + nt) * 512 + lane * 16);
            acc[nt] = __builtin_amdgcn_wmma_f32_16x16x32_bf16(false, a, false, bf,
                                                              (short)0, acc[nt], false, false);
        }
    }
    int mbase = mblk * 64 + wm * 16 + ((lane < 16) ? 0 : 8);
    #pragma unroll
    for (int nt = 0; nt < 4; ++nt) {
        int n = nblk * 128 + wnc * 64 + nt * 16 + (lane & 15);
        int q = n / NW, s = n - q * NW;
        #pragma unroll
        for (int rr = 0; rr < 8; ++rr)
            out[(((size_t)b * NC + mbase + rr) * NOH + (3 * q + r)) * NOH + (3 * s + tt)]
                = acc[nt][rr] * (1.0f / 6.0f);
    }
}

// ================================================================
// host launcher
// ================================================================
extern "C" void kernel_launch(void* const* d_in, const int* in_sizes, int n_in,
                              void* d_out, int out_size, void* d_ws, size_t ws_size,
                              hipStream_t stream) {
    const float* x  = (const float*)d_in[0];
    const float* w1 = (const float*)d_in[1];
    const float* b1 = (const float*)d_in[2];
    const float* a1 = (const float*)d_in[3];
    const float* w2 = (const float*)d_in[4];
    const float* b2 = (const float*)d_in[5];
    const float* a2 = (const float*)d_in[6];
    const float* wa = (const float*)d_in[7];
    const float* ba = (const float*)d_in[8];
    const float* aa = (const float*)d_in[9];
    float* out = (float*)d_out;

    char* ws = (char*)d_ws;
    const size_t SZ_XBT   = (size_t)NB * NHW * NC * 2;
    const size_t SZ_WA    = (size_t)NC * NC * 2;
    const size_t SZ_W1    = (size_t)NCM * NC * 2;
    const size_t SZ_W2    = (size_t)NCM * NC * 2;
    const size_t SZ_EMBED = (size_t)NB * NC  * NHW * 2;
    const size_t SZ_MATCH = (size_t)NB * NCM * NHW * 2;
    const size_t SZ_REF   = (size_t)NB * NCM * 256 * 4;
    const size_t SZ_WN    = (size_t)NB * NL  * NK  * 2;
    const size_t SZ_XPT   = (size_t)NB * NHW * NK  * 2;
    const size_t SZ_SCORE = (size_t)NB * NL  * NHW * 4;
    const size_t SZ_SCT   = (size_t)NB * NHW * NL  * 2;
    size_t off = 0;
    __bf16* xbT   = (__bf16*)(ws + off); off += SZ_XBT;
    __bf16* waz   = (__bf16*)(ws + off); off += SZ_WA;
    __bf16* w1z   = (__bf16*)(ws + off); off += SZ_W1;
    __bf16* w2z   = (__bf16*)(ws + off); off += SZ_W2;
    __bf16* embed = (__bf16*)(ws + off); off += SZ_EMBED;
    __bf16* match = (__bf16*)(ws + off); off += SZ_MATCH;
    float*  ref   = (float*) (ws + off); off += SZ_REF;
    __bf16* wn    = (__bf16*)(ws + off); off += SZ_WN;
    __bf16* xpT   = (__bf16*)(ws + off); off += SZ_XPT;
    float*  score = (float*) (ws + off); off += SZ_SCORE;
    __bf16* ScT   = (__bf16*)(ws + off); off += SZ_SCT;
    __bf16* Ep    = (__bf16*)(ws + off);

    const int BLK = 256;
    auto blocks_w = [](long long waves) { return (int)((waves * 32 + 255) / 256); };
    auto blocks_e = [](long long n)     { return (int)((n + 255) / 256); };

    // 0) one-time layout prep: transpose-cast x, swizzle-cast weights
    k_cast_xT<<<blocks_e((long long)NB * NHW * NC), BLK, 0, stream>>>(x, xbT);
    k_wswz<<<blocks_e(NC * NC),  BLK, 0, stream>>>(wa, waz, NC);
    k_wswz<<<blocks_e(NCM * NC), BLK, 0, stream>>>(w1, w1z, NCM);
    k_wswz<<<blocks_e(NCM * NC), BLK, 0, stream>>>(w2, w2z, NCM);

    // 1) embed + match (1x1 convs, all-vector fragment feeds)
    k_conv1x1<<<blocks_w((long long)NB * (NC / 16) * (NHW / 16)), BLK, 0, stream>>>(
        xbT, waz, ba, aa, embed, NC);
    k_conv1x1<<<blocks_w((long long)NB * (NCM / 16) * (NHW / 16)), BLK, 0, stream>>>(
        xbT, w1z, b1, a1, match, NCM);
    // 2) downsample + conv -> ref
    k_ref_conv<<<blocks_w((long long)NB * (NCM / 16) * (256 / 16)), BLK, 0, stream>>>(
        xbT, w2z, b2, a2, ref);
    // 3) normalized reference patches
    k_wn<<<blocks_w((long long)NB * NL), BLK, 0, stream>>>(ref, wn);
    // 4) im2col of match (transposed, K-contiguous)
    k_im2colT<<<blocks_e((long long)NB * NHW * NK), BLK, 0, stream>>>(match, xpT);
    // 5) score GEMM (async-staged, fragment-layout LDS)
    k_score_t<<<NB * 4 * 18, BLK, 0, stream>>>(wn, xpT, score);
    // 6) softmax over L
    k_softmax<<<blocks_w((long long)NB * NHW), BLK, 0, stream>>>(score);
    // 7) collapse attention (9x FLOP reduction), transposed out
    k_collapseT<<<blocks_e((long long)NB * NHW * NL), BLK, 0, stream>>>(score, ScT);
    // 8) repack embed, then 9 GEMMs -> output
    k_epack<<<blocks_e((long long)NB * 9 * NC * NL), BLK, 0, stream>>>(embed, Ep);
    k_tconv_t<<<NB * 9 * 2 * 18, BLK, 0, stream>>>(Ep, ScT, out);
}
